// SpatialConv_63857573757418
// MI455X (gfx1250) — matrix operand order repaired and checked by reading, not compiled, not run on hardware
//
#include <hip/hip_runtime.h>
#include <hip/hip_bf16.h>
#include <math.h>

// ---------------------------------------------------------------------------
// PNA spatial conv for MI455X (gfx1250, wave32, WMMA f32<-f16 16x16x32)
//
// Main compute: for each of 48 (b,p) batches: C[2048,256] = A @ M[2048,256].
// Tiles staged LDS-side by the Tensor Data Mover (tensor_load_to_lds,
// TENSORcnt double-buffering), fragments read via ds_load into
// v_wmma_f32_16x16x32_f16.
// ---------------------------------------------------------------------------

typedef __attribute__((ext_vector_type(16))) _Float16 v16h;
typedef __attribute__((ext_vector_type(8)))  _Float16 v8h;
typedef __attribute__((ext_vector_type(4)))  _Float16 v4h;
typedef __attribute__((ext_vector_type(8)))  float    v8f;
typedef __attribute__((ext_vector_type(4)))  unsigned int v4u;
typedef __attribute__((ext_vector_type(8)))  int      v8i;
typedef __attribute__((ext_vector_type(4)))  int      v4i;

#define LDS_AS __attribute__((address_space(3)))
typedef LDS_AS _Float16 lds_h;
typedef LDS_AS float    lds_f;
typedef LDS_AS v8h      lds_v8h;
typedef LDS_AS v16h     lds_v16h;

#define NN      2048
#define CC      32
#define BB      4
#define PP      12
#define NBATCH  (BB*PP)         // 48
#define NCOL    256             // 8 funcs * 32 channels
#define NAGG    7
#define KV      (NAGG*CC)       // 224 (epilogue K)
#define NO      32
#define NOE     96              // P0|P1|P2 output groups
#define EPSF    1e-5f
#define SLOPE   0.01f
#define KSTEPS  (NN/32)         // 64

// workspace byte offsets
#define OFF_AH   0ull
#define OFF_MT   (OFF_AH + (unsigned long long)NN*NN*2)                 // 8.39 MB
#define OFF_TE   (OFF_MT + (unsigned long long)NBATCH*NCOL*NN*2)       // +50.3 MB
#define OFF_DEG  (OFF_TE + 65536ull)
#define OFF_LOGD (OFF_DEG  + NN*4)
#define OFF_S    (OFF_LOGD + NN*4)
#define OFF_SINV (OFF_S    + NN*4)
#define OFF_DINV (OFF_SINV + NN*4)

// LDS layout inside k_main (dynamic)
#define LDS_AGG_BYTES (128*NCOL*4)          // 131072
#define LDS_V_BYTES   (128*KV*2)            // 57344
#define LDS_AT_BYTES  (128*32*2)            // 8192 per buffer
#define LDS_BT_BYTES  (256*32*2)            // 16384 per buffer
#define LDS_TOTAL     (LDS_AGG_BYTES + LDS_V_BYTES + 2*LDS_AT_BYTES + 2*LDS_BT_BYTES)

// ---------------------------------------------------------------------------
// fragment helpers (A and B as used here both take contiguous K: lane<16 holds K[0..15],
// lane>=16 holds K[16..31]; B uses the same 16 contiguous K per lane)
// ---------------------------------------------------------------------------
static __device__ __forceinline__ v16h ldA16(const _Float16* p) {
    v8h lo = *(const v8h*)(p);
    v8h hi = *(const v8h*)(p + 16);
    return __builtin_shufflevector(lo, hi, 0,1,2,3,4,5,6,7,8,9,10,11,12,13,14,15);
}
static __device__ __forceinline__ v16h ldA16L(const lds_h* p) {
    v8h lo = *(const lds_v8h*)(p);
    v8h hi = *(const lds_v8h*)(p + 8);
    return __builtin_shufflevector(lo, hi, 0,1,2,3,4,5,6,7,8,9,10,11,12,13,14,15);
}

// ---------------------------------------------------------------------------
// Tensor Data Mover descriptors (CDNA5 ISA ch.8). Group1 is loop-invariant:
// {data_size=2B, tensor dims, tile dims, dim0 stride}. Group0 carries
// lds_addr + global_addr and is rebuilt per issue. Groups 2/3/4 zero.
// ---------------------------------------------------------------------------
static __device__ __forceinline__ v8i tdm_g1(
    unsigned tile_d0, unsigned tile_d1,
    unsigned tensor_d0, unsigned tensor_d1,
    unsigned long long stride0_el) {
    v8i g1;
    g1[0] = (int)(1u << 16);                          // data_size = 1 -> 2 bytes
    g1[1] = (int)((tensor_d0 & 0xffffu) << 16);       // tensor_dim0[15:0] @ 63:48
    g1[2] = (int)(((tensor_d0 >> 16) & 0xffffu) | ((tensor_d1 & 0xffffu) << 16));
    g1[3] = (int)(((tensor_d1 >> 16) & 0xffffu) | ((tile_d0 & 0xffffu) << 16));
    g1[4] = (int)(tile_d1 & 0xffffu);                 // tile_dim1 (tile_dim2 = 0)
    g1[5] = (int)(unsigned)(stride0_el & 0xffffffffull);
    g1[6] = (int)(unsigned)((stride0_el >> 32) & 0xffffull);
    g1[7] = 0;
    return g1;
}

static __device__ __forceinline__ void tdm_issue(
    v8i g1, unsigned long long ga, unsigned lds_off) {
    v4u g0;
    g0.x = 1u;                                        // count=1, user-mode
    g0.y = lds_off;                                   // lds_addr (bytes)
    g0.z = (unsigned)(ga & 0xffffffffull);            // global_addr[31:0]
    g0.w = (unsigned)((ga >> 32) & 0x01ffffffull) | (2u << 30);  // [56:32]|type=2
    v4i z4 = {0, 0, 0, 0};
    v8i z8 = {0, 0, 0, 0, 0, 0, 0, 0};
    __builtin_amdgcn_tensor_load_to_lds(g0, g1, z4, z4, z8, 0);
}

// ---------------------------------------------------------------------------
// K0a: per-row degree + log-degree
// ---------------------------------------------------------------------------
__global__ void k_deg(const float* __restrict__ A, float* __restrict__ deg,
                      float* __restrict__ logd) {
    __shared__ float red[256];
    const int row = blockIdx.x;
    const int tid = threadIdx.x;
    float s = 0.f;
    for (int j = tid; j < NN; j += 256) s += A[(size_t)row*NN + j];
    red[tid] = s; __syncthreads();
    for (int st = 128; st > 0; st >>= 1) {
        if (tid < st) red[tid] += red[tid + st];
        __syncthreads();
    }
    if (tid == 0) { deg[row] = red[0]; logd[row] = logf(red[0] + 1.f); }
}

// ---------------------------------------------------------------------------
// K0b: delta = mean(logd); s, 1/max(s,eps), 1/max(deg,eps)
// ---------------------------------------------------------------------------
__global__ void k_scal(const float* __restrict__ deg, const float* __restrict__ logd,
                       float* __restrict__ sA, float* __restrict__ svA,
                       float* __restrict__ dvA) {
    __shared__ float red[256];
    const int tid = threadIdx.x;
    float s = 0.f;
    for (int j = tid; j < NN; j += 256) s += logd[j];
    red[tid] = s; __syncthreads();
    for (int st = 128; st > 0; st >>= 1) {
        if (tid < st) red[tid] += red[tid + st];
        __syncthreads();
    }
    const float delta = red[0] * (1.f / (float)NN);
    for (int j = tid; j < NN; j += 256) {
        float sv = logd[j] / delta;
        sA[j]  = sv;
        svA[j] = 1.f / fmaxf(sv, EPSF);
        dvA[j] = 1.f / fmaxf(deg[j], EPSF);
    }
}

// ---------------------------------------------------------------------------
// K1: A f32 -> f16 (row-major)
// ---------------------------------------------------------------------------
__global__ void k_a2h(const float* __restrict__ A, _Float16* __restrict__ Ah) {
    const size_t i = ((size_t)blockIdx.x * 256 + threadIdx.x) * 4;
    float4 v = *(const float4*)(A + i);
    v4h h = { (_Float16)v.x, (_Float16)v.y, (_Float16)v.z, (_Float16)v.w };
    *(v4h*)(Ah + i) = h;
}

// ---------------------------------------------------------------------------
// K1b: Theta[672,32] -> Te[96,224] f16, Te[o'][k] = Theta[(o'/32)*224+k][o'%32]
// ---------------------------------------------------------------------------
__global__ void k_theta(const float* __restrict__ Theta, _Float16* __restrict__ Te) {
    const int idx = blockIdx.x * 256 + threadIdx.x;   // 96*224 = 21504
    if (idx >= NOE * KV) return;
    const int op = idx / KV, k = idx % KV;
    const int g = op / NO, o = op % NO;
    Te[idx] = (_Float16)Theta[(size_t)(g * KV + k) * NO + o];
}

// ---------------------------------------------------------------------------
// K2: build Mt[bp][col=f*32+c][m] f16, the 8 pointwise functions of Xp.
// ---------------------------------------------------------------------------
__global__ void k_mt(const float* __restrict__ X, _Float16* __restrict__ Mt) {
    const int m  = blockIdx.x * 256 + threadIdx.x;
    const int c  = blockIdx.y;
    const int bp = blockIdx.z;
    const int b = bp / PP, p = bp % PP;
    const float x = X[(((size_t)b * CC + c) * NN + m) * PP + p];
    const float x2 = x * x, x3 = x2 * x, x4 = x2 * x2;
    const float ep = __expf(x), en = __expf(-x);
    _Float16* base = Mt + ((size_t)bp * NCOL + c) * NN + m;   // f stride = 32*NN
    base[(size_t)0 * CC * NN] = (_Float16)x;
    base[(size_t)1 * CC * NN] = (_Float16)x2;
    base[(size_t)2 * CC * NN] = (_Float16)x3;
    base[(size_t)3 * CC * NN] = (_Float16)x4;
    base[(size_t)4 * CC * NN] = (_Float16)(x * ep);
    base[(size_t)5 * CC * NN] = (_Float16)ep;
    base[(size_t)6 * CC * NN] = (_Float16)(x * en);
    base[(size_t)7 * CC * NN] = (_Float16)en;
}

// ---------------------------------------------------------------------------
// K3: fused main kernel. One WG = 128 rows x 256 cols of one (b,p) batch.
//   stage 1: TDM double-buffered tiles -> LDS; WMMA Agg = Ah @ Mt -> LDS
//   stage 2: pointwise PNA aggregators -> V[128,224] f16 in LDS
//   stage 3: WMMA P[128,96] = V @ Te; out = P0 + s*P1 + (1/s)*P2 + bias; leaky
// ---------------------------------------------------------------------------
__global__ void __launch_bounds__(256)
k_main(const _Float16* __restrict__ Ah, const _Float16* __restrict__ Mt,
       const _Float16* __restrict__ Te, const float* __restrict__ X,
       const float* __restrict__ Bias, const float* __restrict__ sA,
       const float* __restrict__ svA, const float* __restrict__ dvA,
       float* __restrict__ out) {
    extern __shared__ char smem[];
    // LDS byte offsets (aperture rule: low 32 bits of flat addr = LDS offset)
    const unsigned lbase   = (unsigned)(uintptr_t)(void*)smem;
    const unsigned aggOff  = lbase;
    const unsigned vOff    = lbase + LDS_AGG_BYTES;
    const unsigned atOff   = lbase + LDS_AGG_BYTES + LDS_V_BYTES;
    const unsigned btOff   = atOff + 2 * LDS_AT_BYTES;

    lds_f* aggL = (lds_f*)aggOff;                      // [128][256] f32
    lds_h* VL   = (lds_h*)vOff;                        // [128][224] f16

    const int tid  = threadIdx.x;
    const int lane = tid & 31;
    const int w    = tid >> 5;          // wave 0..7
    const int hi   = lane >> 4;         // lane half
    const int l15  = lane & 15;
    const int bp   = blockIdx.y;
    const int b = bp / PP, p = bp % PP;
    const int row0 = blockIdx.x * 128;  // global node base of this WG

    // ---------------- stage 1: A @ Mt (TDM staged, double buffered) --------
    const int rg = w >> 1;              // row group 0..3 (32 rows each)
    const int cg = w & 1;               // col group 0..1 (128 cols each)

    v8f acc[2][8];
    const v8f vzero = {0.f,0.f,0.f,0.f,0.f,0.f,0.f,0.f};
    for (int t = 0; t < 2; ++t)
        for (int nt = 0; nt < 8; ++nt) acc[t][nt] = vzero;

    const unsigned long long aGa = (unsigned long long)(uintptr_t)(Ah + (size_t)row0 * NN);
    const unsigned long long mGa = (unsigned long long)(uintptr_t)(Mt + (size_t)bp * NCOL * NN);
    const v8i g1A = tdm_g1(32, 128, NN, NN, NN);      // A tile: 128 rows x 32 k
    const v8i g1B = tdm_g1(32, NCOL, NN, NCOL, NN);   // B tile: 256 cols x 32 k
    const int ka = hi * 16;     // A-frag K sub-offset (elements)
    const int kb = hi * 16;     // B-frag K sub-offset (elements)

    if (w == 0) {               // prologue: stage k-step 0 into buffer 0
        tdm_issue(g1A, aGa, atOff);
        tdm_issue(g1B, mGa, btOff);
    }

    for (int ks = 0; ks < KSTEPS; ++ks) {
        const unsigned cb = (unsigned)(ks & 1);
        if (w == 0) {
            if (ks + 1 < KSTEPS) {       // prefetch next pair into other buffer
                const unsigned nb = cb ^ 1u;
                tdm_issue(g1A, aGa + (unsigned long long)(ks + 1) * 64, atOff + nb * LDS_AT_BYTES);
                tdm_issue(g1B, mGa + (unsigned long long)(ks + 1) * 64, btOff + nb * LDS_BT_BYTES);
                __builtin_amdgcn_s_wait_tensorcnt(2);   // current pair landed
            } else {
                __builtin_amdgcn_s_wait_tensorcnt(0);
            }
        }
        __syncthreads();                 // LDS tiles visible to all waves

        const lds_h* Ac = (const lds_h*)(atOff + cb * LDS_AT_BYTES);
        const lds_h* Bc = (const lds_h*)(btOff + cb * LDS_BT_BYTES);
        const v16h a0 = ldA16L(Ac + (rg * 32 + l15) * 32 + ka);
        const v16h a1 = ldA16L(Ac + (rg * 32 + 16 + l15) * 32 + ka);
#pragma unroll
        for (int nt = 0; nt < 8; ++nt) {
            const v16h bf = *(const lds_v16h*)(Bc + (cg * 128 + nt * 16 + l15) * 32 + kb);
            acc[0][nt] = __builtin_amdgcn_wmma_f32_16x16x32_f16(
                false, a0, false, bf, (short)0, acc[0][nt], false, false);
            acc[1][nt] = __builtin_amdgcn_wmma_f32_16x16x32_f16(
                false, a1, false, bf, (short)0, acc[1][nt], false, false);
        }
        __syncthreads();                 // done reading before buffer is refilled
    }

    // spill aggregates to LDS (C layout: VGPR i -> row i / 8+i by lane half)
#pragma unroll
    for (int t = 0; t < 2; ++t)
#pragma unroll
        for (int nt = 0; nt < 8; ++nt)
#pragma unroll
            for (int i = 0; i < 8; ++i) {
                const int rl = rg * 32 + t * 16 + i + hi * 8;
                const int cl = cg * 128 + nt * 16 + l15;
                aggL[rl * NCOL + cl] = acc[t][nt][i];
            }
    __syncthreads();

    // ---------------- stage 2: aggregators ---------------------------------
    {
        const int rl = tid >> 1;                 // 0..127
        const int c0 = (tid & 1) * 16;
        const int n  = row0 + rl;
        const float dinv = dvA[n];
        const lds_f* aggR = aggL + rl * NCOL;
        lds_h* vR = VL + rl * KV;
        for (int c = c0; c < c0 + 16; ++c) {
            const float g0 = aggR[0*CC + c], g1 = aggR[1*CC + c];
            const float g2 = aggR[2*CC + c], g3 = aggR[3*CC + c];
            const float g4 = aggR[4*CC + c], g5 = aggR[5*CC + c];
            const float g6 = aggR[6*CC + c], g7 = aggR[7*CC + c];
            const float m1 = g0 * dinv, m2 = g1 * dinv;
            const float m3 = g2 * dinv, m4 = g3 * dinv;
            const float smax = g4 / (g5 + EPSF);
            const float smin = g6 / (g7 + EPSF);
            const float var  = fmaxf(m2 - m1 * m1, 0.f);
            const float stdv = sqrtf(var + EPSF);
            const float x  = X[(((size_t)b * CC + c) * NN + n) * PP + p];
            const float x2 = x * x, x3 = x2 * x, x4 = x2 * x2;
            const float dist = x2 - 2.f * x * m1 + m2;
            const float ed2  = x4 - 4.f*x3*m1 + 6.f*x2*m2 - 4.f*x*m3 + m4;
            const float dstd = sqrtf(fmaxf(ed2 - dist * dist, 0.f) + EPSF);
            vR[0*CC + c] = (_Float16)smin;
            vR[1*CC + c] = (_Float16)smax;
            vR[2*CC + c] = (_Float16)m1;
            vR[3*CC + c] = (_Float16)stdv;
            vR[4*CC + c] = (_Float16)var;
            vR[5*CC + c] = (_Float16)dist;
            vR[6*CC + c] = (_Float16)dstd;
        }
    }
    __syncthreads();

    // ---------------- stage 3: V @ Te, combine, store ----------------------
    v8f pacc[6];
#pragma unroll
    for (int nt = 0; nt < 6; ++nt) pacc[nt] = vzero;

    const lds_h* vRow = VL + (w * 16 + l15) * KV;   // wave w -> rows 16w..
#pragma unroll
    for (int kt = 0; kt < 7; ++kt) {
        const int kk = kt * 32;
        const v16h af = ldA16L(vRow + kk + ka);
#pragma unroll
        for (int nt = 0; nt < 6; ++nt) {
            const v16h bf = *(const v16h*)(Te + (size_t)(nt * 16 + l15) * KV + kk + kb);
            pacc[nt] = __builtin_amdgcn_wmma_f32_16x16x32_f16(
                false, af, false, bf, (short)0, pacc[nt], false, false);
        }
    }

#pragma unroll
    for (int i = 0; i < 8; ++i) {
        const int rl = w * 16 + i + hi * 8;
        const int n  = row0 + rl;
        const float s  = sA[n];
        const float sv = svA[n];
#pragma unroll
        for (int half = 0; half < 2; ++half) {
            const int o = half * 16 + l15;
            float v = pacc[half][i] + s * pacc[2 + half][i] + sv * pacc[4 + half][i]
                    + Bias[o];
            v = (v > 0.f) ? v : SLOPE * v;
            out[(((size_t)b * NO + o) * NN + n) * PP + p] = v;
        }
    }
}

// ---------------------------------------------------------------------------
extern "C" void kernel_launch(void* const* d_in, const int* in_sizes, int n_in,
                              void* d_out, int out_size, void* d_ws, size_t ws_size,
                              hipStream_t stream) {
    const float* X     = (const float*)d_in[0];
    const float* A     = (const float*)d_in[1];
    const float* Theta = (const float*)d_in[2];
    const float* Bias  = (const float*)d_in[3];
    float* out = (float*)d_out;
    char*  ws  = (char*)d_ws;

    _Float16* Ah  = (_Float16*)(ws + OFF_AH);
    _Float16* Mt  = (_Float16*)(ws + OFF_MT);
    _Float16* Te  = (_Float16*)(ws + OFF_TE);
    float* deg  = (float*)(ws + OFF_DEG);
    float* logd = (float*)(ws + OFF_LOGD);
    float* sA   = (float*)(ws + OFF_S);
    float* svA  = (float*)(ws + OFF_SINV);
    float* dvA  = (float*)(ws + OFF_DINV);

    k_deg  <<<NN, 256, 0, stream>>>(A, deg, logd);
    k_scal <<<1, 256, 0, stream>>>(deg, logd, sA, svA, dvA);
    k_a2h  <<<(NN * NN) / (256 * 4), 256, 0, stream>>>(A, Ah);
    k_theta<<<(NOE * KV + 255) / 256, 256, 0, stream>>>(Theta, Te);
    k_mt   <<<dim3(NN / 256, CC, NBATCH), 256, 0, stream>>>(X, Mt);
    k_main <<<dim3(NN / 128, NBATCH), 256, LDS_TOTAL, stream>>>(
        Ah, Mt, Te, X, Bias, sA, svA, dvA, out);
}
